// Attention_1254130450754
// MI455X (gfx1250) — matrix operand order repaired and checked by reading, not compiled, not run on hardware
//
#include <hip/hip_runtime.h>

typedef _Float16 v16h __attribute__((ext_vector_type(16)));
typedef _Float16 v8h  __attribute__((ext_vector_type(8)));
typedef float    v8f  __attribute__((ext_vector_type(8)));

#define CAT16(lo, hi) __builtin_shufflevector((lo), (hi), 0,1,2,3,4,5,6,7,8,9,10,11,12,13,14,15)

static __device__ inline v8f wmma16(v16h a, v16h b, v8f c) {
  // v_wmma_f32_16x16x32_f16: (neg_a, A, neg_b, B, c_mod, C, reuse_a, reuse_b)
  return __builtin_amdgcn_wmma_f32_16x16x32_f16(false, a, false, b, (short)0, c, false, false);
}

#define BATCH   2
#define SEQ     4096
#define DIM     512
#define HEADS   8
#define DH      64
#define NQKV    1536          // Wq(512) | Wk(512) | Wv(512) output columns
#define SCALE   0.125f        // 64^-0.5
#define NEGINF  (-3.0e38f)

// ---------------------------------------------------------------------------
// 1) Convert + transpose weights to f16 [N][K] so WMMA B-fragments are
//    contiguous 32-byte loads. Rows 0..511 = Wq cols, 512..1535 = Wkv cols.
// ---------------------------------------------------------------------------
__global__ __launch_bounds__(256) void cvt_w_kernel(
    const float* __restrict__ Wq, const float* __restrict__ Wkv,
    const float* __restrict__ Wout, _Float16* __restrict__ wtqkv,
    _Float16* __restrict__ woutt) {
  int idx = blockIdx.x * 256 + threadIdx.x;
  if (idx < NQKV * DIM) {
    int c = idx >> 9, k = idx & 511;
    float v = (c < DIM) ? Wq[k * DIM + c] : Wkv[k * 1024 + (c - DIM)];
    wtqkv[c * DIM + k] = (_Float16)v;
  } else {
    int j = idx - NQKV * DIM;       // 0..262143
    int c = j >> 9, k = j & 511;
    woutt[c * DIM + k] = (_Float16)Wout[k * DIM + c];
  }
}

// ---------------------------------------------------------------------------
// 2) LayerNorm (biased var, eps=1e-5) + cast to f16. One block per token row.
// ---------------------------------------------------------------------------
__global__ __launch_bounds__(256) void ln_kernel(
    const float* __restrict__ x, const float* __restrict__ gamma,
    _Float16* __restrict__ xn) {
  const int row = blockIdx.x;
  const int t = threadIdx.x;
  const float* xr = x + (size_t)row * DIM;
  float v0 = xr[t], v1 = xr[t + 256];
  float s = v0 + v1, s2 = v0 * v0 + v1 * v1;
#pragma unroll
  for (int off = 16; off > 0; off >>= 1) {
    s  += __shfl_down(s,  off);
    s2 += __shfl_down(s2, off);
  }
  __shared__ float ls[8], ls2[8], stat[2];
  int wv = t >> 5, ln = t & 31;
  if (ln == 0) { ls[wv] = s; ls2[wv] = s2; }
  __syncthreads();
  if (t == 0) {
    float a = 0.f, b = 0.f;
#pragma unroll
    for (int i = 0; i < 8; ++i) { a += ls[i]; b += ls2[i]; }
    float mean = a * (1.f / DIM);
    float var = b * (1.f / DIM) - mean * mean;
    stat[0] = mean;
    stat[1] = rsqrtf(var + 1e-5f);
  }
  __syncthreads();
  float mean = stat[0], rstd = stat[1];
  xn[(size_t)row * DIM + t]       = (_Float16)((v0 - mean) * rstd * gamma[t]);
  xn[(size_t)row * DIM + t + 256] = (_Float16)((v1 - mean) * rstd * gamma[t + 256]);
}

// ---------------------------------------------------------------------------
// 3) QKV GEMM: [8192 x 512] @ Wt[1536 x 512]^T via WMMA f16.
//    Block = 4 waves, each wave: 16x64 tile, 64 WMMA. q scaled by SCALE on
//    store; v stored TRANSPOSED [bh, d, n] so attention PV B-frags are
//    contiguous.
// ---------------------------------------------------------------------------
__global__ __launch_bounds__(128) void gemm_qkv_kernel(
    const _Float16* __restrict__ xn, const _Float16* __restrict__ wt,
    _Float16* __restrict__ qp, _Float16* __restrict__ kp,
    _Float16* __restrict__ vtp) {
  const int tid = threadIdx.x;
  const int w = tid >> 5, lane = tid & 31, lanelo = lane & 15, hl = lane >> 4;
  const int m0 = blockIdx.x * 64 + w * 16;
  const int n0 = blockIdx.y * 64;
  const int arow = m0 + lanelo;

  v8f acc[4];
#pragma unroll
  for (int t = 0; t < 4; ++t) acc[t] = {};

  for (int k0 = 0; k0 < DIM; k0 += 32) {
    const v8h* ap = (const v8h*)(xn + arow * DIM + k0 + hl * 16);
    v16h a = CAT16(ap[0], ap[1]);   // A: 16 contig K
#pragma unroll
    for (int t = 0; t < 4; ++t) {
      int c = n0 + t * 16 + lanelo;
      v16h b = *(const v16h*)(wt + c * DIM + k0 + hl * 16);  // B: 16 contig K
      acc[t] = wmma16(a, b, acc[t]);
    }
  }

#pragma unroll
  for (int t = 0; t < 4; ++t) {
#pragma unroll
    for (int r = 0; r < 8; ++r) {
      int m = m0 + hl * 8 + r;          // 0..8191
      int c = n0 + t * 16 + lanelo;     // 0..1535
      int bi = m >> 12, i = m & 4095;
      float v = acc[t][r];
      if (c < 512) {
        int h = c >> 6, d = c & 63;
        qp[((bi * HEADS + h) * SEQ + i) * DH + d] = (_Float16)(v * SCALE);
      } else if (c < 1024) {
        int c2 = c - 512, h = c2 >> 6, d = c2 & 63;
        kp[((bi * HEADS + h) * SEQ + i) * DH + d] = (_Float16)v;
      } else {
        int c2 = c - 1024, h = c2 >> 6, d = c2 & 63;
        vtp[((bi * HEADS + h) * DH + d) * SEQ + i] = (_Float16)v;   // V^T
      }
    }
  }
}

// ---------------------------------------------------------------------------
// 4) Flash attention, causal. Block = (bh, 64-query tile), 4 waves x 16 rows.
//    K-tile (8KB) and V^T-tile (8KB) are DMA'd into LDS once per key tile
//    with GLOBAL_LOAD_ASYNC_TO_LDS_B128 (ASYNCcnt), then all 4 waves read
//    WMMA fragments from LDS (broadcast). Online softmax in registers
//    (shfl_xor over the 16-lane D-fragment row groups), P re-laid-out D->A
//    through per-wave LDS, O += P V via WMMA.
// ---------------------------------------------------------------------------
__global__ __launch_bounds__(128) void attn_kernel(
    const _Float16* __restrict__ qp, const _Float16* __restrict__ kp,
    const _Float16* __restrict__ vtp, const unsigned char* __restrict__ maskp,
    _Float16* __restrict__ aop) {
  __shared__ __attribute__((aligned(32))) _Float16 lds_k[64][64];   // [j][d]
  __shared__ __attribute__((aligned(32))) _Float16 lds_v[64][64];   // [d][j]
  __shared__ __attribute__((aligned(32))) _Float16 lds_p[4][16][64];

  const int tid = threadIdx.x;
  const int w = tid >> 5, lane = tid & 31, lanelo = lane & 15, hl = lane >> 4;
  const int bh = blockIdx.y;
  const int bi = bh >> 3, h = bh & 7;
  const int q0 = blockIdx.x * 64;
  const int qi_lane = q0 + w * 16 + lanelo;   // A-fragment row for this lane

  // LDS byte offsets (generic shared pointer low 32 bits == LDS address)
  const unsigned int kbase = (unsigned int)(size_t)&lds_k[0][0];
  const unsigned int vbase = (unsigned int)(size_t)&lds_v[0][0];
  const int crow = tid >> 3;            // 0..15 : row within 16-row chunk
  const int cbyte = (tid & 7) * 16;     // byte column within 128B row

  // Q fragments (d 0..31, 32..63), q pre-scaled by SCALE
  const v8h* qv = (const v8h*)(qp + (bh * SEQ + qi_lane) * DH + hl * 16);
  v16h qa0 = CAT16(qv[0], qv[1]);
  v16h qa1 = CAT16(qv[4], qv[5]);

  float mrow[8], lrow[8];
  v8f o[4];
#pragma unroll
  for (int r = 0; r < 8; ++r) { mrow[r] = NEGINF; lrow[r] = 0.f; }
#pragma unroll
  for (int dc = 0; dc < 4; ++dc) o[dc] = {};

  const unsigned char* mk = maskp + bi * SEQ;

  for (int nt = 0; nt <= blockIdx.x; ++nt) {   // causal: only tiles <= q tile
    const int j0 = nt * 64;

    // ---- async DMA: K tile [j0..j0+63][0..63] and V^T tile into LDS ----
#pragma unroll
    for (int it = 0; it < 4; ++it) {
      int row = it * 16 + crow;                                   // 0..63
      unsigned int kgoff = (unsigned int)(((bh * SEQ + (j0 + row)) * DH) * 2 + cbyte);
      unsigned int kloff = kbase + (unsigned int)(row * 128 + cbyte);
      asm volatile("global_load_async_to_lds_b128 %0, %1, %2"
                   :: "v"(kloff), "v"(kgoff), "s"(kp) : "memory");
      unsigned int vgoff = (unsigned int)(((bh * DH + row) * SEQ + j0) * 2 + cbyte);
      unsigned int vloff = vbase + (unsigned int)(row * 128 + cbyte);
      asm volatile("global_load_async_to_lds_b128 %0, %1, %2"
                   :: "v"(vloff), "v"(vgoff), "s"(vtp) : "memory");
    }
    asm volatile("s_wait_asynccnt 0" ::: "memory");
    __syncthreads();

    // ---- S = Q K^T (16x64 strip per wave, 8 WMMA, K frags from LDS) ----
    v8f s[4];
#pragma unroll
    for (int jc = 0; jc < 4; ++jc) {
      int j = j0 + jc * 16 + lanelo;
      v16h kb0 = *(const v16h*)&lds_k[jc * 16 + lanelo][hl * 16];
      v16h kb1 = *(const v16h*)&lds_k[jc * 16 + lanelo][32 + hl * 16];
      v8f acc = {};
      acc = wmma16(qa0, kb0, acc);
      acc = wmma16(qa1, kb1, acc);
      bool ok = mk[j] != 0;
#pragma unroll
      for (int r = 0; r < 8; ++r) {
        int qi = q0 + w * 16 + hl * 8 + r;
        s[jc][r] = (ok && j <= qi) ? acc[r] : NEGINF;
      }
    }

    // ---- online softmax (per-row stats via 16-lane shfl_xor tree) ----
    float alpha[8];
#pragma unroll
    for (int r = 0; r < 8; ++r) {
      float rm = fmaxf(fmaxf(s[0][r], s[1][r]), fmaxf(s[2][r], s[3][r]));
      rm = fmaxf(rm, __shfl_xor(rm, 1));
      rm = fmaxf(rm, __shfl_xor(rm, 2));
      rm = fmaxf(rm, __shfl_xor(rm, 4));
      rm = fmaxf(rm, __shfl_xor(rm, 8));
      float mn = fmaxf(mrow[r], rm);
      alpha[r] = __expf(mrow[r] - mn);
      mrow[r] = mn;
      float rs = 0.f;
#pragma unroll
      for (int jc = 0; jc < 4; ++jc) {
        float p = __expf(s[jc][r] - mn);
        s[jc][r] = p;
        rs += p;
      }
      rs += __shfl_xor(rs, 1);
      rs += __shfl_xor(rs, 2);
      rs += __shfl_xor(rs, 4);
      rs += __shfl_xor(rs, 8);
      lrow[r] = lrow[r] * alpha[r] + rs;
    }
#pragma unroll
    for (int dc = 0; dc < 4; ++dc)
#pragma unroll
      for (int r = 0; r < 8; ++r) o[dc][r] *= alpha[r];

    // ---- re-layout P: D-fragment -> LDS -> A-fragment (f16) ----
#pragma unroll
    for (int jc = 0; jc < 4; ++jc)
#pragma unroll
      for (int r = 0; r < 8; ++r)
        lds_p[w][hl * 8 + r][jc * 16 + lanelo] = (_Float16)s[jc][r];
    __syncthreads();

    v16h pa0, pa1;
    {
      const v8h* p0 = (const v8h*)&lds_p[w][lanelo][hl * 16];
      pa0 = CAT16(p0[0], p0[1]);          // j {0..15}/{16..31}
      const v8h* p1 = (const v8h*)&lds_p[w][lanelo][32 + hl * 16];
      pa1 = CAT16(p1[0], p1[1]);          // j 32..63
    }

    // ---- O += P V (V^T tile in LDS => broadcast B-frags, 8 WMMA) ----
#pragma unroll
    for (int dc = 0; dc < 4; ++dc) {
      v16h vb0 = *(const v16h*)&lds_v[dc * 16 + lanelo][hl * 16];
      v16h vb1 = *(const v16h*)&lds_v[dc * 16 + lanelo][32 + hl * 16];
      o[dc] = wmma16(pa0, vb0, o[dc]);
      o[dc] = wmma16(pa1, vb1, o[dc]);
    }
    __syncthreads();   // protect lds_k/lds_v/lds_p before next tile's DMA
  }

  // epilogue: normalize and store as [b, n, h*64+d] f16
#pragma unroll
  for (int dc = 0; dc < 4; ++dc)
#pragma unroll
    for (int r = 0; r < 8; ++r) {
      int tok = q0 + w * 16 + hl * 8 + r;
      int col = h * DH + dc * 16 + lanelo;
      aop[((size_t)(bi * SEQ + tok)) * DIM + col] = (_Float16)(o[dc][r] / lrow[r]);
    }
}

// ---------------------------------------------------------------------------
// 5) Output projection: attn_out[8192x512] @ WoutT -> f32 d_out.
// ---------------------------------------------------------------------------
__global__ __launch_bounds__(128) void gemm_out_kernel(
    const _Float16* __restrict__ ao, const _Float16* __restrict__ wt,
    float* __restrict__ outp) {
  const int tid = threadIdx.x;
  const int w = tid >> 5, lane = tid & 31, lanelo = lane & 15, hl = lane >> 4;
  const int m0 = blockIdx.x * 64 + w * 16;
  const int n0 = blockIdx.y * 64;
  const int arow = m0 + lanelo;

  v8f acc[4];
#pragma unroll
  for (int t = 0; t < 4; ++t) acc[t] = {};

  for (int k0 = 0; k0 < DIM; k0 += 32) {
    const v8h* ap = (const v8h*)(ao + arow * DIM + k0 + hl * 16);
    v16h a = CAT16(ap[0], ap[1]);
#pragma unroll
    for (int t = 0; t < 4; ++t) {
      int c = n0 + t * 16 + lanelo;
      v16h b = *(const v16h*)(wt + c * DIM + k0 + hl * 16);
      acc[t] = wmma16(a, b, acc[t]);
    }
  }
#pragma unroll
  for (int t = 0; t < 4; ++t)
#pragma unroll
    for (int r = 0; r < 8; ++r) {
      int m = m0 + hl * 8 + r;
      int c = n0 + t * 16 + lanelo;
      outp[(size_t)m * DIM + c] = acc[t][r];
    }
}

// ---------------------------------------------------------------------------
extern "C" void kernel_launch(void* const* d_in, const int* in_sizes, int n_in,
                              void* d_out, int out_size, void* d_ws, size_t ws_size,
                              hipStream_t stream) {
  const float* x            = (const float*)d_in[0];
  const unsigned char* mask = (const unsigned char*)d_in[1];
  const float* gamma        = (const float*)d_in[2];
  const float* Wq           = (const float*)d_in[3];
  const float* Wkv          = (const float*)d_in[4];
  const float* Wout         = (const float*)d_in[5];
  float* out = (float*)d_out;

  _Float16* ws    = (_Float16*)d_ws;
  _Float16* xn    = ws;                       // 8192*512
  _Float16* wtqkv = xn    + 8192 * 512;       // 1536*512
  _Float16* woutt = wtqkv + 1536 * 512;       // 512*512
  _Float16* q     = woutt + 512 * 512;        // 16*4096*64
  _Float16* k     = q     + 16 * 4096 * 64;
  _Float16* vt    = k     + 16 * 4096 * 64;   // transposed [bh, d, n]
  _Float16* aoph  = vt    + 16 * 4096 * 64;   // 8192*512

  cvt_w_kernel<<<(1536 * 512 + 512 * 512) / 256, 256, 0, stream>>>(
      Wq, Wkv, Wout, wtqkv, woutt);
  ln_kernel<<<8192, 256, 0, stream>>>(x, gamma, xn);
  gemm_qkv_kernel<<<dim3(128, 24), 128, 0, stream>>>(xn, wtqkv, q, k, vt);
  attn_kernel<<<dim3(64, 16), 128, 0, stream>>>(q, k, vt, mask, aoph);
  gemm_out_kernel<<<dim3(128, 8), 128, 0, stream>>>(aoph, woutt, out);
}
